// RotationAwareAttention_52587579572720
// MI455X (gfx1250) — matrix operand order repaired and checked by reading, not compiled, not run on hardware
//
#include <hip/hip_runtime.h>
#include <hip/hip_bf16.h>
#include <math.h>

// ---------------------------------------------------------------------------
// RotationAwareAttention for MI455X (gfx1250, wave32, WMMA).
//
// Algebraic fact: softmax attention is permutation-equivariant, and the rot90
// branches are token permutations composed with pointwise (1x1) convs, so all
// four rotation responses are identical; max == r0.  One attention pass:
// GN -> QKV GEMM -> flash attention -> proj + bias + residual.
//
// Data movement: row-major tiles (weights, Q, K) staged global->LDS with
// GLOBAL_LOAD_ASYNC_TO_LDS_B128 (ASYNCcnt); operands needing transposition
// (x/att GEMM-B tiles, V) staged N-major so every WMMA fragment load is
// 2 x ds_load_b128.  All matrix math: v_wmma_f32_16x16x32_bf16.
// ---------------------------------------------------------------------------

typedef __bf16 bf16;
typedef __attribute__((ext_vector_type(16))) __bf16 v16bf;
typedef __attribute__((ext_vector_type(8)))  float  v8f;

#define BATCH 4
#define CH    256
#define NTOK  1024
#define HEADS 8
#define HDIM  32
#define QK_SCALE 0.17677669529663687f   // (C/HEADS)^-0.5

__device__ __forceinline__ bf16 f2bf(float f) { return static_cast<bf16>(f); }

__device__ __forceinline__ v8f wmma_bf16(v16bf a, v16bf b, v8f c) {
  return __builtin_amdgcn_wmma_f32_16x16x32_bf16(false, a, false, b,
                                                 (short)0, c, false, false);
}

// Async 16B global->LDS copy (gfx1250 ASYNCcnt path, no VGPR round-trip).
__device__ __forceinline__ void async_copy16(const void* gptr, void* sptr) {
  unsigned lds = (unsigned)(size_t)sptr;    // low 32 bits of flat = LDS offset
  asm volatile("global_load_async_to_lds_b128 %0, %1, off"
               :: "v"(lds), "v"(gptr) : "memory");
}
__device__ __forceinline__ void wait_async0() {
  asm volatile("s_wait_asynccnt 0x0" ::: "memory");
}

// A-matrix fragment (16x32 bf16, MxK), element (m,k) at base[m*ld + k].
// Lane = m + 16*khalf; K pairs 2p + 16*khalf -> 16 contiguous
// elements per lane => 2 x ds_load_b128.
__device__ __forceinline__ v16bf load_frag_a(const bf16* base, int ld) {
  const int lane = threadIdx.x & 31;
  const int m = lane & 15, kh = lane >> 4;
  const bf16* row = base + m * ld + 16 * kh;
  v16bf f;
#pragma unroll
  for (int p = 0; p < 8; ++p) {
    const int k = 2 * p;
    f[2 * p]     = row[k];
    f[2 * p + 1] = row[k + 1];
  }
  return f;
}

// B-matrix fragment (32x16 bf16, KxN) from N-major storage:
// element (k,n) = base[n*ld + k].  Lane = n + 16*khalf reads 16 contiguous
// K values => 2 x ds_load_b128.
__device__ __forceinline__ v16bf load_frag_b_t(const bf16* base, int ld) {
  const int lane = threadIdx.x & 31;
  const int n = lane & 15, kh = lane >> 4;
  const bf16* row = base + n * ld + 16 * kh;
  v16bf f;
#pragma unroll
  for (int j = 0; j < 8; ++j) {
    f[2 * j]     = row[2 * j];
    f[2 * j + 1] = row[2 * j + 1];
  }
  return f;
}

// ---------------------------------------------------------------------------
// Kernel 1: f32 -> bf16 weight conversion
// ---------------------------------------------------------------------------
__global__ void cvt_bf16_kernel(const float* __restrict__ src,
                                bf16* __restrict__ dst, int n) {
  int i = blockIdx.x * 256 + threadIdx.x;
  if (i < n) dst[i] = f2bf(src[i]);
}

// ---------------------------------------------------------------------------
// Kernel 2: GroupNorm (32 groups) -> bf16 channel-major (B, C, N)
// ---------------------------------------------------------------------------
__global__ void __launch_bounds__(256)
groupnorm_kernel(const float* __restrict__ x, const float* __restrict__ gamma,
                 const float* __restrict__ beta, bf16* __restrict__ xn) {
  const int blk = blockIdx.x;          // b*32 + g
  const int b = blk >> 5, g = blk & 31;
  const int t = threadIdx.x;
  const float* base = x + (size_t)((b * CH + g * 8) << 10);
  float s = 0.f, s2 = 0.f;
#pragma unroll
  for (int i = 0; i < 32; ++i) {
    float v = base[t + (i << 8)];
    s += v; s2 += v * v;
  }
  __shared__ float red[256], red2[256];
  __shared__ float s_mu, s_rs;
  red[t] = s; red2[t] = s2;
  __syncthreads();
  for (int off = 128; off > 0; off >>= 1) {
    if (t < off) { red[t] += red[t + off]; red2[t] += red2[t + off]; }
    __syncthreads();
  }
  if (t == 0) {
    float mu  = red[0] * (1.0f / 8192.0f);
    float var = red2[0] * (1.0f / 8192.0f) - mu * mu;
    s_mu = mu; s_rs = rsqrtf(var + 1e-5f);
  }
  __syncthreads();
  const float mu = s_mu, rs = s_rs;
  bf16* out = xn + (size_t)((b * CH + g * 8) << 10);
#pragma unroll
  for (int i = 0; i < 32; ++i) {
    int e = t + (i << 8);
    int c = g * 8 + (e >> 10);
    out[e] = f2bf((base[e] - mu) * rs * gamma[c] + beta[c]);
  }
}

// ---------------------------------------------------------------------------
// Kernel 3: QKV GEMM.  qkv[b,o,n] = sum_c w[o,c] * xn[b,c,n]
// WG tile 128(o) x 64(n), 8 waves x (16x64).  A staged async; B staged
// transposed (n-major) for vectorized fragment loads.  Q outputs pre-scaled.
// ---------------------------------------------------------------------------
__global__ void __launch_bounds__(256)
qkv_gemm_kernel(const bf16* __restrict__ wq,    // 768 x 256 row-major
                const bf16* __restrict__ xn,    // (B, 256, 1024)
                bf16* __restrict__ q, bf16* __restrict__ k,
                bf16* __restrict__ v) {
  const int blk = blockIdx.x;                   // 4 * 6 * 16
  const int b = blk / 96, rem = blk % 96;
  const int mt = rem >> 4, nt = rem & 15;
  const int o0 = mt * 128, n0 = nt * 64;
  const int t = threadIdx.x, wave = t >> 5, lane = t & 31;
  const int hh = lane >> 4, nl = lane & 15;

  __shared__ bf16 sA[128 * 32];   // w tile, o-major, ld 32
  __shared__ bf16 sBt[64 * 32];   // x tile, n-major,  ld 32

  v8f zero = {};
  v8f acc[4] = {zero, zero, zero, zero};

  for (int kk = 0; kk < CH; kk += 32) {
    // ---- async stage A: 128x32, 512 x 16B chunks ----
#pragma unroll
    for (int i = 0; i < 2; ++i) {
      int ch = t + i * 256;
      int r = ch >> 2, seg = ch & 3;
      async_copy16(&wq[(o0 + r) * CH + kk + seg * 8], &sA[r * 32 + seg * 8]);
    }
    // ---- transpose-stage B: read 8 n-contiguous, write n-major ----
    {
      int kc = t >> 3, nseg = t & 7;
      union { uint4 u; bf16 h[8]; } raw;
      raw.u = *(const uint4*)(&xn[(size_t)(b * CH + kk + kc) * NTOK + n0 + nseg * 8]);
#pragma unroll
      for (int e = 0; e < 8; ++e)
        sBt[(nseg * 8 + e) * 32 + kc] = raw.h[e];
    }
    wait_async0();
    __syncthreads();
    v16bf a = load_frag_a(&sA[(wave * 16) * 32], 32);
#pragma unroll
    for (int j = 0; j < 4; ++j) {
      v16bf bb = load_frag_b_t(&sBt[(j * 16) * 32], 32);
      acc[j] = wmma_bf16(a, bb, acc[j]);
    }
    __syncthreads();
  }
  // Scatter D tiles to token-major q/k/v; fold softmax scale into Q.
  const int o_base = o0 + wave * 16;
  const int sect = o_base >> 8;                 // uniform per block
  bf16* dst = (sect == 0) ? q : ((sect == 1) ? k : v);
  const float oscale = (sect == 0) ? QK_SCALE : 1.0f;
#pragma unroll
  for (int j = 0; j < 4; ++j) {
#pragma unroll
    for (int r = 0; r < 8; ++r) {
      int o = o_base + r + 8 * hh;
      int n = n0 + j * 16 + nl;
      int c = o & 255;
      int h = c >> 5, d = c & 31;
      dst[((size_t)(b * HEADS + h) * NTOK + n) * HDIM + d] =
          f2bf(acc[j][r] * oscale);
    }
  }
}

// ---------------------------------------------------------------------------
// Kernel 4: flash attention, one WG per (b, h, 128-row q tile).
// Q/K staged async (token-major feeds vectorized A / B^T fragments);
// V staged transposed (d-major).  8 WMMA for S, 8 WMMA for P*V per kv tile.
// ---------------------------------------------------------------------------
__global__ void __launch_bounds__(256)
flash_attn_kernel(const bf16* __restrict__ qm, const bf16* __restrict__ km,
                  const bf16* __restrict__ vm, bf16* __restrict__ att) {
  const int blk = blockIdx.x;                   // (b*8+h)*8 + qt
  const int qt = blk & 7, bh = blk >> 3;
  const int b = bh >> 3, h = bh & 7;
  const int q0 = qt * 128;
  const int t = threadIdx.x, wave = t >> 5, lane = t & 31;
  const int hh = lane >> 4, nl = lane & 15;

  __shared__ bf16 sQ[128 * 32];                 // token-major
  __shared__ bf16 sK[128 * 32];                 // token-major
  __shared__ bf16 sVt[32 * 128];                // d-major (transposed)
  __shared__ bf16 sP[8][16 * 128];              // per-wave P staging

  const bf16* qb = qm + ((size_t)bh * NTOK + q0) * HDIM;
#pragma unroll
  for (int i = 0; i < 2; ++i) {
    int ch = t + i * 256;
    async_copy16(&qb[ch * 8], &sQ[ch * 8]);
  }
  wait_async0();
  __syncthreads();
  v16bf a_q = load_frag_a(&sQ[(wave * 16) * 32], 32);

  v8f zero = {};
  v8f o_acc[2] = {zero, zero};                  // 16 x 32 f32
  float m_run[8], l_run[8];
#pragma unroll
  for (int r = 0; r < 8; ++r) { m_run[r] = -1e30f; l_run[r] = 0.f; }

  for (int kt = 0; kt < 8; ++kt) {
    const bf16* kb = km + ((size_t)bh * NTOK + kt * 128) * HDIM;
    const bf16* vb = vm + ((size_t)bh * NTOK + kt * 128) * HDIM;
    // async stage K (token-major), prefetch next tile
#pragma unroll
    for (int i = 0; i < 2; ++i) {
      int ch = t + i * 256;
      async_copy16(&kb[ch * 8], &sK[ch * 8]);
      if (kt < 7) {
        __builtin_prefetch(&kb[4096 + ch * 8], 0, 0);   // global_prefetch_b8
        __builtin_prefetch(&vb[4096 + ch * 8], 0, 0);
      }
    }
    // transpose-stage V: sVt[d][token]
#pragma unroll
    for (int i = 0; i < 2; ++i) {
      int ch = t + i * 256;
      int tok = ch >> 2, dseg = ch & 3;
      union { uint4 u; bf16 h[8]; } raw;
      raw.u = *(const uint4*)(&vb[tok * HDIM + dseg * 8]);
#pragma unroll
      for (int e = 0; e < 8; ++e)
        sVt[(dseg * 8 + e) * 128 + tok] = raw.h[e];
    }
    wait_async0();
    __syncthreads();

    // ---- S strip: this wave's 16 q-rows x 128 kv-cols (Q pre-scaled) ----
    v8f s[8];
#pragma unroll
    for (int tt = 0; tt < 8; ++tt) {
      v16bf b_k = load_frag_b_t(&sK[(tt * 16) * 32], 32);  // K^T fragment
      s[tt] = wmma_bf16(a_q, b_k, zero);
    }
    float rm[8];
#pragma unroll
    for (int r = 0; r < 8; ++r) rm[r] = -1e30f;
#pragma unroll
    for (int tt = 0; tt < 8; ++tt)
#pragma unroll
      for (int r = 0; r < 8; ++r) rm[r] = fmaxf(rm[r], s[tt][r]);
    // row-max across the 16-lane half owning each row (xor<=8 stays in-half)
#pragma unroll
    for (int r = 0; r < 8; ++r)
#pragma unroll
      for (int msk = 8; msk >= 1; msk >>= 1)
        rm[r] = fmaxf(rm[r], __shfl_xor(rm[r], msk, 32));

    float sf[8];
#pragma unroll
    for (int r = 0; r < 8; ++r) {
      float mn = fmaxf(m_run[r], rm[r]);
      sf[r] = __expf(m_run[r] - mn);
      m_run[r] = mn;
      l_run[r] *= sf[r];
    }
#pragma unroll
    for (int j = 0; j < 2; ++j)
#pragma unroll
      for (int r = 0; r < 8; ++r) o_acc[j][r] *= sf[r];

    // exp + row-sum + restage P (bf16, row-major 16x128)
    float rs[8];
#pragma unroll
    for (int r = 0; r < 8; ++r) rs[r] = 0.f;
    bf16* pw = &sP[wave][0];
#pragma unroll
    for (int tt = 0; tt < 8; ++tt)
#pragma unroll
      for (int r = 0; r < 8; ++r) {
        float p = __expf(s[tt][r] - m_run[r]);
        rs[r] += p;
        pw[(r + 8 * hh) * 128 + tt * 16 + nl] = f2bf(p);
      }
#pragma unroll
    for (int r = 0; r < 8; ++r) {
#pragma unroll
      for (int msk = 8; msk >= 1; msk >>= 1)
        rs[r] += __shfl_xor(rs[r], msk, 32);
      l_run[r] += rs[r];
    }

    // ---- O += P * V : K=128 -> 4 A fragments x 2 d-tiles ----
#pragma unroll
    for (int kk = 0; kk < 4; ++kk) {
      v16bf a_p = load_frag_a(pw + kk * 32, 128);
#pragma unroll
      for (int j = 0; j < 2; ++j) {
        v16bf b_v = load_frag_b_t(&sVt[(j * 16) * 128 + kk * 32], 128);
        o_acc[j] = wmma_bf16(a_p, b_v, o_acc[j]);
      }
    }
    __syncthreads();
  }

  // normalize and store channel-major (b, h*32+d, n)
#pragma unroll
  for (int j = 0; j < 2; ++j)
#pragma unroll
    for (int r = 0; r < 8; ++r) {
      int n = q0 + wave * 16 + r + 8 * hh;
      int d = j * 16 + nl;
      att[((size_t)(b * CH + h * HDIM + d) << 10) + n] =
          f2bf(o_acc[j][r] / l_run[r]);
    }
}

// ---------------------------------------------------------------------------
// Kernel 5: projection GEMM + bias + residual (f32 output, coalesced).
// ---------------------------------------------------------------------------
__global__ void __launch_bounds__(256)
proj_kernel(const bf16* __restrict__ wp,        // 256 x 256 row-major
            const bf16* __restrict__ att,       // (B, 256, 1024)
            const float* __restrict__ bproj, const float* __restrict__ x,
            float* __restrict__ out) {
  const int blk = blockIdx.x;                   // 4 * 2 * 16
  const int b = blk >> 5, rem = blk & 31;
  const int mt = rem >> 4, nt = rem & 15;
  const int o0 = mt * 128, n0 = nt * 64;
  const int t = threadIdx.x, wave = t >> 5, lane = t & 31;
  const int hh = lane >> 4, nl = lane & 15;

  __shared__ bf16 sA[128 * 32];
  __shared__ bf16 sBt[64 * 32];
  v8f zero = {};
  v8f acc[4] = {zero, zero, zero, zero};

  for (int kk = 0; kk < CH; kk += 32) {
#pragma unroll
    for (int i = 0; i < 2; ++i) {
      int ch = t + i * 256;
      int r = ch >> 2, seg = ch & 3;
      async_copy16(&wp[(o0 + r) * CH + kk + seg * 8], &sA[r * 32 + seg * 8]);
    }
    {
      int kc = t >> 3, nseg = t & 7;
      union { uint4 u; bf16 h[8]; } raw;
      raw.u = *(const uint4*)(&att[(size_t)(b * CH + kk + kc) * NTOK + n0 + nseg * 8]);
#pragma unroll
      for (int e = 0; e < 8; ++e)
        sBt[(nseg * 8 + e) * 32 + kc] = raw.h[e];
    }
    wait_async0();
    __syncthreads();
    v16bf a = load_frag_a(&sA[(wave * 16) * 32], 32);
#pragma unroll
    for (int j = 0; j < 4; ++j) {
      v16bf bb = load_frag_b_t(&sBt[(j * 16) * 32], 32);
      acc[j] = wmma_bf16(a, bb, acc[j]);
    }
    __syncthreads();
  }
#pragma unroll
  for (int j = 0; j < 4; ++j)
#pragma unroll
    for (int r = 0; r < 8; ++r) {
      int o = o0 + wave * 16 + r + 8 * hh;
      int n = n0 + j * 16 + nl;
      size_t idx = ((size_t)(b * CH + o) << 10) + n;
      out[idx] = acc[j][r] + bproj[o] + x[idx];
    }
}

// ---------------------------------------------------------------------------
extern "C" void kernel_launch(void* const* d_in, const int* in_sizes, int n_in,
                              void* d_out, int out_size, void* d_ws,
                              size_t ws_size, hipStream_t stream) {
  const float* x     = (const float*)d_in[0];   // (4,256,32,32)
  const float* gamma = (const float*)d_in[1];   // (256,)
  const float* beta  = (const float*)d_in[2];   // (256,)
  const float* wqkv  = (const float*)d_in[3];   // (768,256)
  const float* wproj = (const float*)d_in[4];   // (256,256)
  const float* bproj = (const float*)d_in[5];   // (256,)
  float* out = (float*)d_out;

  char* ws = (char*)d_ws;
  size_t off = 0;
  auto alloc = [&](size_t bytes) -> void* {
    void* p = ws + off;
    off = (off + bytes + 255) & ~(size_t)255;
    return p;
  };
  bf16* xn      = (bf16*)alloc((size_t)BATCH * CH * NTOK * 2);
  bf16* wqkv_bf = (bf16*)alloc((size_t)3 * CH * CH * 2);
  bf16* wproj_bf= (bf16*)alloc((size_t)CH * CH * 2);
  bf16* q_ws    = (bf16*)alloc((size_t)BATCH * HEADS * NTOK * HDIM * 2);
  bf16* k_ws    = (bf16*)alloc((size_t)BATCH * HEADS * NTOK * HDIM * 2);
  bf16* v_ws    = (bf16*)alloc((size_t)BATCH * HEADS * NTOK * HDIM * 2);
  bf16* att_ws  = (bf16*)alloc((size_t)BATCH * CH * NTOK * 2);

  cvt_bf16_kernel<<<(3 * CH * CH + 255) / 256, 256, 0, stream>>>(wqkv, wqkv_bf,
                                                                 3 * CH * CH);
  cvt_bf16_kernel<<<(CH * CH + 255) / 256, 256, 0, stream>>>(wproj, wproj_bf,
                                                             CH * CH);
  groupnorm_kernel<<<BATCH * 32, 256, 0, stream>>>(x, gamma, beta, xn);
  qkv_gemm_kernel<<<BATCH * 6 * 16, 256, 0, stream>>>(wqkv_bf, xn, q_ws, k_ws,
                                                      v_ws);
  flash_attn_kernel<<<BATCH * HEADS * 8, 256, 0, stream>>>(q_ws, k_ws, v_ws,
                                                           att_ws);
  proj_kernel<<<BATCH * 2 * 16, 256, 0, stream>>>(wproj_bf, att_ws, bproj, x,
                                                  out);
}
